// Evolution_4664334483942
// MI455X (gfx1250) — matrix operand order repaired and checked by reading, not compiled, not run on hardware
//
#include <hip/hip_runtime.h>
#include <hip/hip_bf16.h>
#include <math.h>

#define DD 512      // hidden dim
#define GG 2048     // 4*D gates
#define RR 8        // relations
#define MAXB 1024   // max batch size
#define LDT 40      // padded LDS row stride in halves: 80B rows -> 16B aligned,
                    // and 20*n mod 64 distinct for n<16 -> conflict-free frag loads
#define LDC 136     // C-tile LDS row stride in halves (272B, 16B aligned)

typedef __attribute__((ext_vector_type(16))) _Float16 v16h;
typedef __attribute__((ext_vector_type(4)))  _Float16 v4h;
typedef __attribute__((ext_vector_type(2)))  _Float16 h2;
typedef __attribute__((ext_vector_type(8)))  float    v8f;

union FragU { v16h v; unsigned u[8]; };

// ---- CDNA5 async global->LDS copy (ASYNCcnt path), 16B per lane ----
__device__ __forceinline__ void async_ld_b128(void* lds_dst, const void* gsrc) {
  const unsigned lds_off = (unsigned)(unsigned long long)(uintptr_t)lds_dst;
  asm volatile("global_load_async_to_lds_b128 %0, %1, off"
               :: "v"(lds_off), "v"(gsrc) : "memory");
}
__device__ __forceinline__ void wait_async0() {
  asm volatile("s_wait_asynccnt 0x0" ::: "memory");
}

// A fragment 16x32 (MxK), 16-bit: lanes 0-15: m=lane, K={0..7,16..23}; lanes 16-31: +8
__device__ __forceinline__ v16h frag_a(const _Float16* base, int lane) {
  const int m = lane & 15;
  const _Float16* p = base + m * LDT + ((lane & 16) ? 8 : 0);
  FragU r;
#pragma unroll
  for (int i = 0; i < 4; ++i) {
    r.u[i]     = *(const unsigned*)(p + 2 * i);
    r.u[4 + i] = *(const unsigned*)(p + 16 + 2 * i);
  }
  return r.v;
}

// B fragment 32x16 (KxN), staged transposed in LDS as Bs[n][k]:
// lanes 0-15: n=lane, K={0..7,16..23}; lanes 16-31: n=lane-16, K={8..15,24..31}
__device__ __forceinline__ v16h frag_b(const _Float16* base, int lane) {
  const int n = lane & 15;
  const _Float16* p = base + n * LDT + ((lane & 16) ? 8 : 0);
  FragU r;
#pragma unroll
  for (int i = 0; i < 4; ++i) {
    r.u[i]     = *(const unsigned*)(p + 2 * i);
    r.u[4 + i] = *(const unsigned*)(p + 16 + 2 * i);
  }
  return r.v;
}

__device__ __forceinline__ v8f wmma_f16(v16h a, v16h b, v8f c) {
  return __builtin_amdgcn_wmma_f32_16x16x32_f16(false, a, false, b, (short)0, c, false, false);
}

// ---------------- setup kernels ----------------

__global__ void k_init(float* c, unsigned* hbf_u, int* cnt, int* offsets,
                       const int* __restrict__ batch_sizes, int T) {
  const int i = blockIdx.x * blockDim.x + threadIdx.x;
  if (i < MAXB * DD) c[i] = 0.0f;
  if (i < MAXB * DD / 2) hbf_u[i] = 0u;
  if (i < RR) cnt[i] = 0;
  if (i == 0) {
    int s = 0;
    for (int t = 0; t < T; ++t) { offsets[t] = s; s += batch_sizes[t]; }
  }
}

__global__ void k_hist(const int* __restrict__ rels, int* cnt, int n) {
  const int i = blockIdx.x * blockDim.x + threadIdx.x;
  if (i < n) atomicAdd(&cnt[rels[i]], 1);
}

__global__ void k_scan(const int* __restrict__ cnt, int* bst, int* cur) {
  int s = 0;
  for (int r = 0; r < RR; ++r) { bst[r] = s; cur[r] = s; s += cnt[r]; }
}

__global__ void k_scatter(const int* __restrict__ rels, int* cur,
                          int* __restrict__ bucket, int n) {
  const int i = blockIdx.x * blockDim.x + threadIdx.x;
  if (i < n) { const int p = atomicAdd(&cur[rels[i]], 1); bucket[p] = i; }
}

__global__ void k_whh(const float* __restrict__ W_hh, _Float16* __restrict__ Whh) {
  const int i = blockIdx.x * blockDim.x + threadIdx.x;
  if (i < GG * DD) Whh[i] = (_Float16)W_hh[i];
}

// Permuted combined bias: bcomb[r][np*4+g] = W_ih[g*512+np,:].b_rel[r,:] + b_ih + b_hh
__global__ void k_bias(const float* __restrict__ W_ih, const float* __restrict__ b_rel,
                       const float* __restrict__ b_ih, const float* __restrict__ b_hh,
                       float* __restrict__ bcomb) {
  const int i = blockIdx.x * blockDim.x + threadIdx.x;
  if (i >= RR * GG) return;
  const int r = i / GG, jp = i % GG;
  const int jorig = (jp & 3) * DD + (jp >> 2);   // gate-interleaved -> original column
  const float* w  = W_ih + (size_t)jorig * DD;
  const float* br = b_rel + (size_t)r * DD;
  float s = b_ih[jorig] + b_hh[jorig];
  for (int k = 0; k < DD; ++k) s += w[k] * br[k];
  bcomb[i] = s;
}

// ---------------- M[r] = W_ih(2048x512) @ W_rel[r](512x512) -> f16 ----------------

__global__ __launch_bounds__(256) void k_gemm_prep(const float* __restrict__ W_ih,
                                                   const float* __restrict__ W_rel,
                                                   _Float16* __restrict__ M) {
  __shared__ _Float16 As[128 * LDT];
  __shared__ _Float16 Bs[128 * LDT];
  const int r  = blockIdx.z;
  const int m0 = blockIdx.y * 128;   // over 2048 (j)
  const int n0 = blockIdx.x * 128;   // over 512  (k of M)
  const int tid = threadIdx.x, lane = tid & 31, wid = tid >> 5;
  const int wm = wid >> 2, wn = wid & 3;   // 2 x 4 waves, each 64 x 32
  const float* Wr = W_rel + (size_t)r * DD * DD;

  v8f acc[4][2];
#pragma unroll
  for (int i = 0; i < 4; ++i)
#pragma unroll
    for (int j = 0; j < 2; ++j)
#pragma unroll
      for (int v = 0; v < 8; ++v) acc[i][j][v] = 0.0f;

  for (int kk = 0; kk < DD; kk += 32) {
    { // stage A: W_ih rows, f32 -> f16
      const int row = tid >> 1, half = tid & 1;
      const float4* src = (const float4*)(W_ih + (size_t)(m0 + row) * DD + kk + half * 16);
      _Float16* dst = As + row * LDT + half * 16;
#pragma unroll
      for (int q = 0; q < 4; ++q) {
        const float4 f = src[q];
        h2 lo = { (_Float16)f.x, (_Float16)f.y };
        h2 hi = { (_Float16)f.z, (_Float16)f.w };
        *(h2*)(dst + q * 4)     = lo;
        *(h2*)(dst + q * 4 + 2) = hi;
      }
    }
    { // stage B transposed: Bs[n][k2] = W_rel[r][kk+k2][n0+n]
      const int k2 = tid >> 3, seg = tid & 7;
      const float4* src = (const float4*)(Wr + (size_t)(kk + k2) * DD + n0 + seg * 16);
#pragma unroll
      for (int q = 0; q < 4; ++q) {
        const float4 f = src[q];
        const int nb = seg * 16 + q * 4;
        Bs[(nb + 0) * LDT + k2] = (_Float16)f.x;
        Bs[(nb + 1) * LDT + k2] = (_Float16)f.y;
        Bs[(nb + 2) * LDT + k2] = (_Float16)f.z;
        Bs[(nb + 3) * LDT + k2] = (_Float16)f.w;
      }
    }
    __syncthreads();
    v16h af[4], bf[2];
#pragma unroll
    for (int i = 0; i < 4; ++i) af[i] = frag_a(As + (wm * 64 + i * 16) * LDT, lane);
#pragma unroll
    for (int j = 0; j < 2; ++j) bf[j] = frag_b(Bs + (wn * 32 + j * 16) * LDT, lane);
#pragma unroll
    for (int i = 0; i < 4; ++i)
#pragma unroll
      for (int j = 0; j < 2; ++j) acc[i][j] = wmma_f16(af[i], bf[j], acc[i][j]);
    __syncthreads();
  }
  const int hi8 = (lane & 16) ? 8 : 0, ln = lane & 15;
#pragma unroll
  for (int i = 0; i < 4; ++i)
#pragma unroll
    for (int j = 0; j < 2; ++j)
#pragma unroll
      for (int v = 0; v < 8; ++v) {
        const int gm = m0 + wm * 64 + i * 16 + v + hi8;
        const int gn = n0 + wn * 32 + j * 16 + ln;
        M[((size_t)r * GG + gm) * DD + gn] = (_Float16)acc[i][j][v];
      }
}

// ------------- gates_x: embed[nodes[tok]] @ M[r].T + b_comb, written gate-interleaved -------------
// gx[tok][np*4+g]  (np in [0,512), g in 0..3)

__device__ __forceinline__ void stage_a_gates(_Float16* As, const float* __restrict__ embed,
                                              const int* nds, int kk, int tid) {
  const int row = tid >> 1, half = tid & 1;
  const int nd = nds[row];
  _Float16* dst = As + row * LDT + half * 16;
  if (nd >= 0) {
    const float4* src = (const float4*)(embed + (size_t)nd * DD + kk + half * 16);
#pragma unroll
    for (int q = 0; q < 4; ++q) {
      const float4 f = src[q];
      h2 lo = { (_Float16)f.x, (_Float16)f.y };
      h2 hi = { (_Float16)f.z, (_Float16)f.w };
      *(h2*)(dst + q * 4)     = lo;
      *(h2*)(dst + q * 4 + 2) = hi;
    }
  } else {
    h2 z = { (_Float16)0.0f, (_Float16)0.0f };
#pragma unroll
    for (int q = 0; q < 8; ++q) *(h2*)(dst + q * 2) = z;
  }
}

// tile column cc corresponds to interleaved index jp = jp0+cc; original M row = (jp&3)*512 + (jp>>2)
__device__ __forceinline__ void issue_b_gates(_Float16* Bs, const _Float16* __restrict__ Mr,
                                              int jp0, int kk, int tid) {
  const int cc = tid >> 1, half = tid & 1;
  const int jp = jp0 + cc;
  const int jorig = (jp & 3) * DD + (jp >> 2);
  const _Float16* src = Mr + (size_t)jorig * DD + kk + half * 16; // 32B, 16B-aligned
  _Float16* dst = Bs + cc * LDT + half * 16;                      // 16B-aligned (LDT=40)
  async_ld_b128(dst, src);
  async_ld_b128(dst + 8, src + 8);
}

__global__ __launch_bounds__(256) void k_gemm_gates(const float* __restrict__ embed,
                                                    const int* __restrict__ nodes,
                                                    const int* __restrict__ bucket,
                                                    const int* __restrict__ bst,
                                                    const int* __restrict__ bcnt,
                                                    const _Float16* __restrict__ M,
                                                    const float* __restrict__ bcomb,
                                                    _Float16* __restrict__ gx) {
  __shared__ _Float16 smem[4 * 128 * LDT];   // 2x A buf + 2x B buf; reused as C tile
  __shared__ int toks[128];
  __shared__ int nds[128];
  _Float16* const As0 = smem;
  _Float16* const As1 = smem + 128 * LDT;
  _Float16* const Bs0 = smem + 2 * 128 * LDT;
  _Float16* const Bs1 = smem + 3 * 128 * LDT;

  const int r  = blockIdx.z;
  const int nr = bcnt[r];
  const int row0 = blockIdx.y * 128;
  if (row0 >= nr) return;
  const int jp0 = blockIdx.x * 128;   // contiguous block in gate-interleaved space
  const int tid = threadIdx.x, lane = tid & 31, wid = tid >> 5;
  const int wm = wid >> 2, wn = wid & 3;
  const int base = bst[r];
  if (tid < 128) {
    int tk = -1;
    if (row0 + tid < nr) tk = bucket[base + row0 + tid];
    toks[tid] = tk;
    nds[tid] = (tk >= 0) ? nodes[tk] : -1;
  }
  __syncthreads();

  v8f acc[4][2];
#pragma unroll
  for (int i = 0; i < 4; ++i)
#pragma unroll
    for (int j = 0; j < 2; ++j)
#pragma unroll
      for (int v = 0; v < 8; ++v) acc[i][j][v] = 0.0f;

  const _Float16* Mr = M + (size_t)r * GG * DD;

  // prologue: fill buffer 0
  stage_a_gates(As0, embed, nds, 0, tid);
  issue_b_gates(Bs0, Mr, jp0, 0, tid);
  wait_async0();
  __syncthreads();

  int pb = 0;
  for (int kk = 0; kk < DD; kk += 32, pb ^= 1) {
    const int nxt = kk + 32;
    _Float16* Ac = pb ? As1 : As0;
    _Float16* Bc = pb ? Bs1 : Bs0;
    if (nxt < DD) {              // software pipeline: fill other buffer
      stage_a_gates(pb ? As0 : As1, embed, nds, nxt, tid);
      issue_b_gates(pb ? Bs0 : Bs1, Mr, jp0, nxt, tid);
    }
    v16h af[4], bf[2];
#pragma unroll
    for (int i = 0; i < 4; ++i) af[i] = frag_a(Ac + (wm * 64 + i * 16) * LDT, lane);
#pragma unroll
    for (int j = 0; j < 2; ++j) bf[j] = frag_b(Bc + (wn * 32 + j * 16) * LDT, lane);
#pragma unroll
    for (int i = 0; i < 4; ++i)
#pragma unroll
      for (int j = 0; j < 2; ++j) acc[i][j] = wmma_f16(af[i], bf[j], acc[i][j]);
    wait_async0();
    __syncthreads();
  }

  // ---- LDS-staged epilogue: bias add + f16 convert into C tile, then coalesced stores ----
  _Float16* const ctile = smem;   // 128 x LDC halves = 34.8KB, smem is free now
  const int hi8 = (lane & 16) ? 8 : 0, ln = lane & 15;
#pragma unroll
  for (int i = 0; i < 4; ++i)
#pragma unroll
    for (int j = 0; j < 2; ++j) {
      const int ccol = wn * 32 + j * 16 + ln;
      const float bb = bcomb[(size_t)r * GG + jp0 + ccol];
#pragma unroll
      for (int v = 0; v < 8; ++v) {
        const int rr_ = wm * 64 + i * 16 + v + hi8;
        ctile[rr_ * LDC + ccol] = (_Float16)(acc[i][j][v] + bb);
      }
    }
  __syncthreads();
  {
    const int row = tid >> 1, half = tid & 1;
    const int tk = toks[row];
    if (tk >= 0) {
      const uint4* srcc = (const uint4*)(ctile + row * LDC + half * 64); // 272B row stride: aligned
      uint4* dstg = (uint4*)(gx + (size_t)tk * GG + jp0 + half * 64);    // 128B contiguous
#pragma unroll
      for (int q = 0; q < 8; ++q) dstg[q] = srcc[q];
    }
  }
}

// ------------- one LSTM time step: gates = gx + h @ W_hh.T, then elementwise update -------------

__device__ __forceinline__ void issue_a_step(_Float16* As, const _Float16* __restrict__ hbf,
                                             int b0, int kk, int tid) {
  const int row = tid >> 2, q = tid & 3;
  async_ld_b128(As + row * LDT + q * 8, hbf + (size_t)(b0 + row) * DD + kk + q * 8);
}

__device__ __forceinline__ void issue_b_step(_Float16* Bs, const _Float16* __restrict__ Whh,
                                             int n0, int kk, int tid) {
  const int g = tid >> 6, cl = tid & 63;
  const _Float16* src = Whh + (size_t)(g * DD + n0 + cl) * DD + kk;  // row j = g*512+n0+cl
  _Float16* dst = Bs + (g * 64 + cl) * LDT;
#pragma unroll
  for (int q = 0; q < 4; ++q) async_ld_b128(dst + q * 8, src + q * 8);
}

__global__ __launch_bounds__(256) void k_step(const _Float16* __restrict__ gx,
                                              const _Float16* __restrict__ Whh,
                                              _Float16* __restrict__ hbf,
                                              float* __restrict__ c,
                                              float* __restrict__ out,
                                              const int* __restrict__ batch_sizes,
                                              const int* __restrict__ offsets, int t) {
  __shared__ _Float16 As[2][64 * LDT];    // h rows
  __shared__ _Float16 Bs[2][256 * LDT];   // 4 gates x 64 cols
  const int bs = batch_sizes[t];
  const int b0 = blockIdx.y * 64;
  if (b0 >= bs) return;
  const int off = offsets[t];
  const int n0 = blockIdx.x * 64;      // hidden-col tile within [0,512)
  const int tid = threadIdx.x, lane = tid & 31, wid = tid >> 5;
  const int wm = wid >> 1, wn = wid & 1;  // 4 x 2 waves: 16 rows x 32 cols each

  v8f acc[4][2];  // [gate][n-frag]
#pragma unroll
  for (int g = 0; g < 4; ++g)
#pragma unroll
    for (int j = 0; j < 2; ++j)
#pragma unroll
      for (int v = 0; v < 8; ++v) acc[g][j][v] = 0.0f;

  issue_a_step(As[0], hbf, b0, 0, tid);
  issue_b_step(Bs[0], Whh, n0, 0, tid);
  wait_async0();
  __syncthreads();

  int pb = 0;
  for (int kk = 0; kk < DD; kk += 32, pb ^= 1) {
    const int nxt = kk + 32;
    if (nxt < DD) {
      issue_a_step(As[pb ^ 1], hbf, b0, nxt, tid);
      issue_b_step(Bs[pb ^ 1], Whh, n0, nxt, tid);
    }
    const v16h a = frag_a(As[pb] + (wm * 16) * LDT, lane);
#pragma unroll
    for (int g = 0; g < 4; ++g)
#pragma unroll
      for (int j = 0; j < 2; ++j) {
        const v16h b = frag_b(Bs[pb] + (g * 64 + wn * 32 + j * 16) * LDT, lane);
        acc[g][j] = wmma_f16(a, b, acc[g][j]);
      }
    wait_async0();
    __syncthreads();
  }

  const int hi8 = (lane & 16) ? 8 : 0, ln = lane & 15;
#pragma unroll
  for (int j = 0; j < 2; ++j)
#pragma unroll
    for (int v = 0; v < 8; ++v) {
      const int b = b0 + wm * 16 + v + hi8;
      if (b < bs) {
        const int np = n0 + wn * 32 + j * 16 + ln;
        // one 8B load fetches all four gate values (gate-interleaved gx layout)
        const v4h gq = *(const v4h*)(gx + (size_t)(off + b) * GG + np * 4);
        const float iv = acc[0][j][v] + (float)gq[0];
        const float fv = acc[1][j][v] + (float)gq[1];
        const float gv = acc[2][j][v] + (float)gq[2];
        const float ov = acc[3][j][v] + (float)gq[3];
        const float si = 1.0f / (1.0f + __expf(-iv));
        const float sf = 1.0f / (1.0f + __expf(-fv));
        const float so = 1.0f / (1.0f + __expf(-ov));
        const float cv = c[(size_t)b * DD + np];
        const float cn = sf * cv + si * tanhf(gv);
        const float hn = so * tanhf(cn);
        c[(size_t)b * DD + np] = cn;
        hbf[(size_t)b * DD + np] = (_Float16)hn;
        out[(size_t)(off + b) * DD + np] = hn;
      }
    }
}

// ---------------- host driver ----------------

extern "C" void kernel_launch(void* const* d_in, const int* in_sizes, int n_in,
                              void* d_out, int out_size, void* d_ws, size_t ws_size,
                              hipStream_t stream) {
  const float* embed = (const float*)d_in[0];
  const float* W_rel = (const float*)d_in[1];
  const float* b_rel = (const float*)d_in[2];
  const float* W_ih  = (const float*)d_in[3];
  const float* W_hh  = (const float*)d_in[4];
  const float* b_ih  = (const float*)d_in[5];
  const float* b_hh  = (const float*)d_in[6];
  const int* nodes = (const int*)d_in[7];
  const int* rels  = (const int*)d_in[8];
  const int* batch_sizes = (const int*)d_in[11];
  const int n_tok = in_sizes[7];
  const int T = in_sizes[11];
  float* out = (float*)d_out;

  char* ws = (char*)d_ws;
  size_t o = 0;
  auto carve = [&](size_t bytes) -> void* {
    void* p = (void*)(ws + o);
    o += (bytes + 255) & ~(size_t)255;
    return p;
  };
  int* offsets   = (int*)carve((size_t)T * sizeof(int));
  int* cnt       = (int*)carve(RR * sizeof(int));
  int* bst       = (int*)carve(RR * sizeof(int));
  int* cur       = (int*)carve(RR * sizeof(int));
  int* bucket    = (int*)carve((size_t)n_tok * sizeof(int));
  float* bcomb   = (float*)carve((size_t)RR * GG * sizeof(float));
  _Float16* Whh  = (_Float16*)carve((size_t)GG * DD * sizeof(_Float16));
  _Float16* M    = (_Float16*)carve((size_t)RR * GG * DD * sizeof(_Float16));
  _Float16* hbf  = (_Float16*)carve((size_t)MAXB * DD * sizeof(_Float16));
  float* cst     = (float*)carve((size_t)MAXB * DD * sizeof(float));
  _Float16* gx   = (_Float16*)carve((size_t)n_tok * GG * sizeof(_Float16));

  k_init<<<dim3((MAXB * DD + 255) / 256), 256, 0, stream>>>(cst, (unsigned*)hbf, cnt, offsets, batch_sizes, T);
  k_hist<<<dim3((n_tok + 255) / 256), 256, 0, stream>>>(rels, cnt, n_tok);
  k_scan<<<1, 1, 0, stream>>>(cnt, bst, cur);
  k_scatter<<<dim3((n_tok + 255) / 256), 256, 0, stream>>>(rels, cur, bucket, n_tok);
  k_whh<<<dim3((GG * DD + 255) / 256), 256, 0, stream>>>(W_hh, Whh);
  k_bias<<<dim3((RR * GG + 255) / 256), 256, 0, stream>>>(W_ih, b_rel, b_ih, b_hh, bcomb);
  k_gemm_prep<<<dim3(DD / 128, GG / 128, RR), 256, 0, stream>>>(W_ih, W_rel, M);
  k_gemm_gates<<<dim3(GG / 128, (n_tok + 127) / 128, RR), 256, 0, stream>>>(
      embed, nodes, bucket, bst, cnt, M, bcomb, gx);
  for (int t = 0; t < T; ++t)
    k_step<<<dim3(DD / 64, MAXB / 64), 256, 0, stream>>>(gx, Whh, hbf, cst, out, batch_sizes, offsets, t);
}
